// Decoder_3032246911551
// MI455X (gfx1250) — matrix operand order repaired and checked by reading, not compiled, not run on hardware
//
#include <hip/hip_runtime.h>
#include <hip/hip_bf16.h>

// ---------------------------------------------------------------------------
// Transformer decoder forward for MI455X (gfx1250, wave32).
// All GEMMs run on v_wmma_f32_16x16x32_bf16. Tiles are staged into LDS
// *pre-swizzled into WMMA fragment order* (fragment load = 2x ds_load_b128),
// with double-buffered LDS and software-pipelined global loads. Out-of-range
// staging uses clamped addresses (no exec-mask predication); the epilogue
// guards all stores, so garbage rows/columns are never written back.
// ---------------------------------------------------------------------------

typedef __attribute__((ext_vector_type(16))) __bf16 bf16x16;
typedef __attribute__((ext_vector_type(8)))  __bf16 bf16x8;
typedef __attribute__((ext_vector_type(2)))  __bf16 bf16x2;
typedef __attribute__((ext_vector_type(8)))  float  f32x8;

#define BM 64
#define BN 128
#define BK 32
#define NTHREADS 256

__device__ __forceinline__ unsigned pack_bf16(float lo, float hi) {
    bf16x2 t;
    t[0] = (__bf16)lo;
    t[1] = (__bf16)hi;
    return __builtin_bit_cast(unsigned, t);
}

// Two contiguous 16B LDS reads -> one 16-element bf16 fragment in VGPRs.
__device__ __forceinline__ bf16x16 ld_frag(const __bf16* p) {
    const bf16x8* q = reinterpret_cast<const bf16x8*>(p);
    bf16x8 lo = q[0];
    bf16x8 hi = q[1];
    return __builtin_shufflevector(lo, hi, 0, 1, 2, 3, 4, 5, 6, 7,
                                   8, 9, 10, 11, 12, 13, 14, 15);
}

// Generic batched GEMM:  C[m,n] = epilogue( sum_k A[m,k] * B[k,n] )
//   A[m,k] = A[m*lda + k]                       (k contiguous, always)
//   B[k,n] = BTRANS ? B[n*ldb + k] : B[k*ldb + n]
// Batch (grid.z): z -> zo=z/zdiv, zi=z%zdiv; per-operand offsets applied.
// op: 0 = none, 1 = +bias, 2 = relu(+bias), 3 = mask/scale (attention scores)
//
// LDS is stored in WMMA wave32 fragment order:
//   Aswz[t][lane][e] : t = m-subtile (16 rows), lane 0..31, e 0..15
//        row = t*16 + lane%16 ; k = 16*(lane/16) + e
//   Bswz[u][lane][e] : u = n-subtile (16 cols)
//        col = u*16 + lane%16 ; k = 16*(lane/16) + e
template <int BTRANS>
__global__ __launch_bounds__(NTHREADS)
void gemm_bf16_kernel(const float* __restrict__ A, const float* __restrict__ Bm,
                      const float* __restrict__ bias, const int* __restrict__ mask,
                      float* __restrict__ C,
                      int M, int N, int K,
                      long lda, long ldb, long ldc,
                      int op, float scale,
                      int zdiv,
                      long ao, long ai, long bo, long bi, long co, long ci)
{
    __shared__ __bf16 Aswz[2][4][32][16];   //  8 KB, double buffered
    __shared__ __bf16 Bswz[2][8][32][16];   // 16 KB, double buffered

    const int z  = blockIdx.z;
    const int zo = z / zdiv;
    const int zi = z % zdiv;
    A  += (long)zo * ao + (long)zi * ai;
    Bm += (long)zo * bo + (long)zi * bi;
    C  += (long)zo * co + (long)zi * ci;

    const int tid  = threadIdx.x;
    const int lane = tid & 31;
    const int wave = tid >> 5;     // 0..7
    const int wm   = wave >> 2;    // 0..1  (M direction, 32 rows each)
    const int wn   = wave & 3;     // 0..3  (N direction, 32 cols each)
    const int l16  = lane & 15;
    const int g16  = lane >> 4;

    const int m0 = blockIdx.y * BM;
    const int n0 = blockIdx.x * BN;

    // ---- staging decomposition (constant per thread) ----
    // A: this thread loads 8 consecutive k of one row.
    const int a_row  = tid >> 2;              // 0..63
    const int a_col  = (tid & 3) << 3;        // 0,8,16,24
    const int a_t    = a_row >> 4;            // m-subtile
    const int a_lane = ((a_col >> 4) << 4) | (a_row & 15);
    const int a_e0   = a_col & 8;             // 0 or 8
    // B: this thread produces 16 k-values of one column (n).
    const int b_n    = tid & 127;             // 0..127
    const int b_kh   = tid >> 7;              // 0..1 (k half: 16 k each)
    const int b_u    = b_n >> 4;              // n-subtile
    const int b_lane = (b_kh << 4) | (b_n & 15);

    // Clamped source indices: OOB rows/cols only feed C entries the guarded
    // epilogue never writes, so unconditional loads are safe.
    const int a_rowc = min(m0 + a_row, M - 1);
    const int b_nc   = min(n0 + b_n, N - 1);

    const float* a_src = A + (long)a_rowc * lda + a_col;               // + k0
    const float* b_src = BTRANS ? (Bm + (long)b_nc * ldb + (b_kh << 4))   // + k0
                                : (Bm + (long)(b_kh << 4) * ldb + b_nc);  // + k0*ldb

    float4 ra0, ra1;                      // incoming A payload
    float4 rb0, rb1, rb2, rb3;            // incoming B payload (BTRANS)
    float  rbn[16];                       // incoming B payload (normal)

    auto gload = [&](int k0) {
        ra0 = *reinterpret_cast<const float4*>(a_src + k0);
        ra1 = *reinterpret_cast<const float4*>(a_src + k0 + 4);
        if (BTRANS) {
            const float* s = b_src + k0;
            rb0 = *reinterpret_cast<const float4*>(s);
            rb1 = *reinterpret_cast<const float4*>(s + 4);
            rb2 = *reinterpret_cast<const float4*>(s + 8);
            rb3 = *reinterpret_cast<const float4*>(s + 12);
        } else {
            const float* s = b_src + (long)k0 * ldb;   // wave reads 32 consecutive n per k-row
            #pragma unroll
            for (int p = 0; p < 16; ++p) rbn[p] = s[(long)p * ldb];
        }
    };
    auto lstore = [&](int buf) {
        *reinterpret_cast<uint4*>(&Aswz[buf][a_t][a_lane][a_e0]) =
            make_uint4(pack_bf16(ra0.x, ra0.y), pack_bf16(ra0.z, ra0.w),
                       pack_bf16(ra1.x, ra1.y), pack_bf16(ra1.z, ra1.w));
        if (BTRANS) {
            *reinterpret_cast<uint4*>(&Bswz[buf][b_u][b_lane][0]) =
                make_uint4(pack_bf16(rb0.x, rb0.y), pack_bf16(rb0.z, rb0.w),
                           pack_bf16(rb1.x, rb1.y), pack_bf16(rb1.z, rb1.w));
            *reinterpret_cast<uint4*>(&Bswz[buf][b_u][b_lane][8]) =
                make_uint4(pack_bf16(rb2.x, rb2.y), pack_bf16(rb2.z, rb2.w),
                           pack_bf16(rb3.x, rb3.y), pack_bf16(rb3.z, rb3.w));
        } else {
            *reinterpret_cast<uint4*>(&Bswz[buf][b_u][b_lane][0]) =
                make_uint4(pack_bf16(rbn[0], rbn[1]), pack_bf16(rbn[2],  rbn[3]),
                           pack_bf16(rbn[4], rbn[5]), pack_bf16(rbn[6],  rbn[7]));
            *reinterpret_cast<uint4*>(&Bswz[buf][b_u][b_lane][8]) =
                make_uint4(pack_bf16(rbn[8],  rbn[9]),  pack_bf16(rbn[10], rbn[11]),
                           pack_bf16(rbn[12], rbn[13]), pack_bf16(rbn[14], rbn[15]));
        }
    };

    f32x8 acc[2][2];
    #pragma unroll
    for (int i = 0; i < 2; ++i)
        #pragma unroll
        for (int j = 0; j < 2; ++j)
            #pragma unroll
            for (int r = 0; r < 8; ++r) acc[i][j][r] = 0.0f;

    // ---- software pipeline: prologue ----
    gload(0);
    lstore(0);
    __syncthreads();

    int buf = 0;
    for (int k0 = 0; k0 < K; k0 += BK) {
        const bool has_next = (k0 + BK) < K;
        if (has_next) gload(k0 + BK);      // issue global loads before WMMA
        if (k0 + 2 * BK < K) {             // GL2 prefetch two tiles ahead
            __builtin_prefetch(a_src + k0 + 2 * BK, 0, 1);
            if (BTRANS) __builtin_prefetch(b_src + k0 + 2 * BK, 0, 1);
            else        __builtin_prefetch(b_src + (long)(k0 + 2 * BK) * ldb, 0, 1);
        }

        // ---- fragment loads: pure ds_load_b128 pairs ----
        bf16x16 af0 = ld_frag(&Aswz[buf][wm * 2 + 0][lane][0]);
        bf16x16 af1 = ld_frag(&Aswz[buf][wm * 2 + 1][lane][0]);
        bf16x16 bf0 = ld_frag(&Bswz[buf][wn * 2 + 0][lane][0]);
        bf16x16 bf1 = ld_frag(&Bswz[buf][wn * 2 + 1][lane][0]);

        acc[0][0] = __builtin_amdgcn_wmma_f32_16x16x32_bf16(
            false, af0, false, bf0, (short)0, acc[0][0], false, false);
        acc[0][1] = __builtin_amdgcn_wmma_f32_16x16x32_bf16(
            false, af0, false, bf1, (short)0, acc[0][1], false, false);
        acc[1][0] = __builtin_amdgcn_wmma_f32_16x16x32_bf16(
            false, af1, false, bf0, (short)0, acc[1][0], false, false);
        acc[1][1] = __builtin_amdgcn_wmma_f32_16x16x32_bf16(
            false, af1, false, bf1, (short)0, acc[1][1], false, false);

        if (has_next) lstore(buf ^ 1);     // convert + store into other buffer
        __syncthreads();
        buf ^= 1;
    }

    // ---- epilogue (f32 C/D layout: VGPR r -> row r + 8*(lane/16), col lane%16)
    #pragma unroll
    for (int i = 0; i < 2; ++i) {
        #pragma unroll
        for (int j = 0; j < 2; ++j) {
            const int colg = n0 + wn * 32 + j * 16 + l16;
            if (colg >= N) continue;
            const float bv = (op == 1 || op == 2) ? bias[colg] : 0.0f;
            #pragma unroll
            for (int r = 0; r < 8; ++r) {
                const int rowg = m0 + wm * 32 + i * 16 + r + g16 * 8;
                if (rowg >= M) continue;
                float f = acc[i][j][r];
                if      (op == 1) { f += bv; }
                else if (op == 2) { f += bv; f = fmaxf(f, 0.0f); }
                else if (op == 3) { f = (mask[(long)rowg * N + colg] == 0) ? -10000.0f : f * scale; }
                C[(long)rowg * ldc + colg] = f;
            }
        }
    }
}

// Row-wise softmax (in place), one block per row.
__global__ __launch_bounds__(256)
void softmax_kernel(float* __restrict__ s, int ncols)
{
    __shared__ float red[256];
    const long base = (long)blockIdx.x * ncols;
    const int tid = threadIdx.x;
    float mx = -3.4e38f;
    for (int c = tid; c < ncols; c += 256) mx = fmaxf(mx, s[base + c]);
    red[tid] = mx; __syncthreads();
    for (int off = 128; off > 0; off >>= 1) {
        if (tid < off) red[tid] = fmaxf(red[tid], red[tid + off]);
        __syncthreads();
    }
    mx = red[0]; __syncthreads();
    float sum = 0.0f;
    for (int c = tid; c < ncols; c += 256) {
        float e = __expf(s[base + c] - mx);
        s[base + c] = e;
        sum += e;
    }
    red[tid] = sum; __syncthreads();
    for (int off = 128; off > 0; off >>= 1) {
        if (tid < off) red[tid] += red[tid + off];
        __syncthreads();
    }
    const float inv = 1.0f / red[0];
    for (int c = tid; c < ncols; c += 256) s[base + c] *= inv;
}

// out = LayerNorm(a + r) * g + b, row length Dd == 1024, one block per row.
// (out may alias r: values are held in registers before the final writes)
__global__ __launch_bounds__(256)
void add_ln_kernel(const float* __restrict__ a, const float* __restrict__ r,
                   const float* __restrict__ g, const float* __restrict__ b,
                   float* __restrict__ out, int Dd)
{
    __shared__ float red[256];
    const long base = (long)blockIdx.x * Dd;
    const int tid = threadIdx.x;
    float v[4];
    float s = 0.0f;
    for (int i = 0; i < 4; ++i) {
        const int c = tid + i * 256;
        v[i] = a[base + c] + r[base + c];
        s += v[i];
    }
    red[tid] = s; __syncthreads();
    for (int off = 128; off > 0; off >>= 1) { if (tid < off) red[tid] += red[tid + off]; __syncthreads(); }
    const float mean = red[0] / (float)Dd; __syncthreads();
    float q = 0.0f;
    for (int i = 0; i < 4; ++i) { const float d = v[i] - mean; q += d * d; }
    red[tid] = q; __syncthreads();
    for (int off = 128; off > 0; off >>= 1) { if (tid < off) red[tid] += red[tid + off]; __syncthreads(); }
    const float inv = rsqrtf(red[0] / (float)Dd + 1e-10f);
    for (int i = 0; i < 4; ++i) {
        const int c = tid + i * 256;
        out[base + c] = g[c] * (v[i] - mean) * inv + b[c];
    }
}

// x[b,s,:] = emb[dec[b,s],:] + positional encoding(s,:)
__global__ __launch_bounds__(256)
void embed_pe_kernel(const int* __restrict__ dec, const float* __restrict__ emb,
                     float* __restrict__ x, int S, int Dd)
{
    const int idx  = blockIdx.x;         // 0..B*S-1
    const int spos = idx % S;
    const int tok  = dec[idx];
    const long ob = (long)idx * Dd;
    const long eb = (long)tok * Dd;
    for (int d = threadIdx.x; d < Dd; d += 256) {
        const int p = d >> 1;
        const float ang = (float)spos * __powf(10000.0f, -(2.0f * (float)p) / (float)Dd);
        const float pe = (d & 1) ? __cosf(ang) : __sinf(ang);
        x[ob + d] = emb[eb + d] + pe;
    }
}

extern "C" void kernel_launch(void* const* d_in, const int* in_sizes, int n_in,
                              void* d_out, int out_size, void* d_ws, size_t ws_size,
                              hipStream_t stream)
{
    const int Bd = 4, S = 512, Dd = 1024, Hh = 16, HD = 64, Ff = 4096, Ll = 6, Vv = 32000;
    const int M = Bd * S;  // 2048 flattened tokens

    const int*   dec   = (const int*)  d_in[0];
    const float* enc   = (const float*)d_in[1];
    const int*   tmask = (const int*)  d_in[2];
    const int*   smask = (const int*)  d_in[3];
    const float* emb   = (const float*)d_in[4];
    const float* sa_wq = (const float*)d_in[5];
    const float* sa_bq = (const float*)d_in[6];
    const float* sa_wk = (const float*)d_in[7];
    const float* sa_bk = (const float*)d_in[8];
    const float* sa_wv = (const float*)d_in[9];
    const float* sa_bv = (const float*)d_in[10];
    const float* sa_wo = (const float*)d_in[11];
    const float* sa_bo = (const float*)d_in[12];
    const float* ca_wq = (const float*)d_in[13];
    const float* ca_bq = (const float*)d_in[14];
    const float* ca_wk = (const float*)d_in[15];
    const float* ca_bk = (const float*)d_in[16];
    const float* ca_wv = (const float*)d_in[17];
    const float* ca_bv = (const float*)d_in[18];
    const float* ca_wo = (const float*)d_in[19];
    const float* ca_bo = (const float*)d_in[20];
    const float* fw1   = (const float*)d_in[21];
    const float* fb1   = (const float*)d_in[22];
    const float* fw2   = (const float*)d_in[23];
    const float* fb2   = (const float*)d_in[24];
    const float* ln1g  = (const float*)d_in[25];
    const float* ln1b  = (const float*)d_in[26];
    const float* ln2g  = (const float*)d_in[27];
    const float* ln2b  = (const float*)d_in[28];
    const float* ln3g  = (const float*)d_in[29];
    const float* ln3b  = (const float*)d_in[30];
    const float* wout  = (const float*)d_in[31];
    const float* bout  = (const float*)d_in[32];
    float* out = (float*)d_out;

    // workspace carve-up (fp32): ~151 MB total
    float* ws = (float*)d_ws;
    const size_t MD = (size_t)M * Dd;
    float* x   = ws;                            // [M, D] running activations
    float* t1  = x   + MD;                      // [M, D] attn / ffn output
    float* qb  = t1  + MD;                      // [M, D]
    float* kb  = qb  + MD;                      // [M, D]
    float* vb  = kb  + MD;                      // [M, D]
    float* ctx = vb  + MD;                      // [M, D]
    float* sc  = ctx + MD;                      // [B*H, S, S] scores/probs
    float* fh  = sc  + (size_t)Bd * Hh * S * S; // [M, F]

    auto gemm = [&](const float* A, const float* Bmat, const float* bias, const int* mask,
                    float* C, int Mm, int Nn, int Kk, long lda, long ldb, long ldc,
                    int btrans, int op, float scale,
                    int Z, int zdiv, long ao, long ai, long bo, long bi, long co, long ci) {
        dim3 grid((Nn + BN - 1) / BN, (Mm + BM - 1) / BM, Z);
        if (btrans)
            gemm_bf16_kernel<1><<<grid, dim3(NTHREADS), 0, stream>>>(
                A, Bmat, bias, mask, C, Mm, Nn, Kk, lda, ldb, ldc,
                op, scale, zdiv, ao, ai, bo, bi, co, ci);
        else
            gemm_bf16_kernel<0><<<grid, dim3(NTHREADS), 0, stream>>>(
                A, Bmat, bias, mask, C, Mm, Nn, Kk, lda, ldb, ldc,
                op, scale, zdiv, ao, ai, bo, bi, co, ci);
    };

    const long SD = (long)S * Dd;
    const long SS = (long)S * S;
    const float inv_sqrt_hd = 0.125f;  // 1/sqrt(64)

    embed_pe_kernel<<<M, 256, 0, stream>>>(dec, emb, x, S, Dd);

    for (int l = 0; l < Ll; ++l) {
        const size_t wO = (size_t)l * Dd * Dd, bO = (size_t)l * Dd;

        // ---------------- self-attention ----------------
        gemm(x, sa_wq + wO, sa_bq + bO, nullptr, qb, M, Dd, Dd, Dd, Dd, Dd, 0, 1, 1.f, 1, 1, 0,0,0,0,0,0);
        gemm(x, sa_wk + wO, sa_bk + bO, nullptr, kb, M, Dd, Dd, Dd, Dd, Dd, 0, 1, 1.f, 1, 1, 0,0,0,0,0,0);
        gemm(x, sa_wv + wO, sa_bv + bO, nullptr, vb, M, Dd, Dd, Dd, Dd, Dd, 0, 1, 1.f, 1, 1, 0,0,0,0,0,0);
        // scores[b,h] = Q[b,:,h*HD:+HD] @ K^T, masked + scaled
        gemm(qb, kb, nullptr, tmask, sc, S, S, HD, Dd, Dd, S, 1, 3, inv_sqrt_hd,
             Bd * Hh, Hh, SD, HD, SD, HD, (long)Hh * SS, SS);
        softmax_kernel<<<Bd * Hh * S, 256, 0, stream>>>(sc, S);
        // ctx[b,:,h*HD:+HD] = probs @ V
        gemm(sc, vb, nullptr, nullptr, ctx, S, HD, S, S, Dd, Dd, 0, 0, 1.f,
             Bd * Hh, Hh, (long)Hh * SS, SS, SD, HD, SD, HD);
        gemm(ctx, sa_wo + wO, sa_bo + bO, nullptr, t1, M, Dd, Dd, Dd, Dd, Dd, 0, 1, 1.f, 1, 1, 0,0,0,0,0,0);
        add_ln_kernel<<<M, 256, 0, stream>>>(t1, x, ln1g + bO, ln1b + bO, x, Dd);

        // ---------------- cross-attention ----------------
        gemm(x,   ca_wq + wO, ca_bq + bO, nullptr, qb, M, Dd, Dd, Dd, Dd, Dd, 0, 1, 1.f, 1, 1, 0,0,0,0,0,0);
        gemm(enc, ca_wk + wO, ca_bk + bO, nullptr, kb, M, Dd, Dd, Dd, Dd, Dd, 0, 1, 1.f, 1, 1, 0,0,0,0,0,0);
        gemm(enc, ca_wv + wO, ca_bv + bO, nullptr, vb, M, Dd, Dd, Dd, Dd, Dd, 0, 1, 1.f, 1, 1, 0,0,0,0,0,0);
        gemm(qb, kb, nullptr, smask, sc, S, S, HD, Dd, Dd, S, 1, 3, inv_sqrt_hd,
             Bd * Hh, Hh, SD, HD, SD, HD, (long)Hh * SS, SS);
        softmax_kernel<<<Bd * Hh * S, 256, 0, stream>>>(sc, S);
        gemm(sc, vb, nullptr, nullptr, ctx, S, HD, S, S, Dd, Dd, 0, 0, 1.f,
             Bd * Hh, Hh, (long)Hh * SS, SS, SD, HD, SD, HD);
        gemm(ctx, ca_wo + wO, ca_bo + bO, nullptr, t1, M, Dd, Dd, Dd, Dd, Dd, 0, 1, 1.f, 1, 1, 0,0,0,0,0,0);
        add_ln_kernel<<<M, 256, 0, stream>>>(t1, x, ln2g + bO, ln2b + bO, x, Dd);

        // ---------------- FFN ----------------
        gemm(x,  fw1 + (size_t)l * Dd * Ff, fb1 + (size_t)l * Ff, nullptr, fh,
             M, Ff, Dd, Dd, Ff, Ff, 0, 2, 1.f, 1, 1, 0,0,0,0,0,0);
        gemm(fh, fw2 + (size_t)l * Ff * Dd, fb2 + bO, nullptr, t1,
             M, Dd, Ff, Ff, Dd, Dd, 0, 1, 1.f, 1, 1, 0,0,0,0,0,0);
        add_ln_kernel<<<M, 256, 0, stream>>>(t1, x, ln3g + bO, ln3b + bO, x, Dd);
    }

    // logits = x @ w_out + b_out   -> d_out [M, V]
    gemm(x, wout, bout, nullptr, out, M, Vv, Dd, Dd, Vv, Vv, 0, 1, 1.f, 1, 1, 0,0,0,0,0,0);
}
